// ConvLSTMModel_13434657702484
// MI455X (gfx1250) — matrix operand order repaired and checked by reading, not compiled, not run on hardware
//
#include <hip/hip_runtime.h>
#include <hip/hip_bf16.h>

// ---------- types ----------
typedef __attribute__((ext_vector_type(16))) __bf16 v16bf;
typedef __attribute__((ext_vector_type(8)))  float  v8f;

#define IN_CH   8
#define HIDDEN  64
#define HORIZON 3
#define BATCH   8
#define TSTEPS  10
#define HH      96
#define WW      96
#define CPAD    96              // 72 real channels padded to 3*32
#define NPIX    (HH * WW)       // 9216
#define HN      (BATCH * HIDDEN * NPIX)   // 4,718,592 state elements

// A_packed: 16 mtiles * 9 taps * 3 kchunks = 432 fragments * 512 halfs
#define NFRAG   (16 * 9 * 3)

// workspace layout (bytes)
#define A_OFF   0u
#define H0_OFF  (1u << 20)                      // 1 MiB
#define H1_OFF  (H0_OFF + 2u * HN)              // h bf16 = 9,437,184 B
#define C_OFF   (H1_OFF + 2u * HN)
// total ~38.8 MB

__device__ __forceinline__ unsigned short f2bf(float f) {
    unsigned u = __float_as_uint(f);
    return (unsigned short)((u + 0x7FFFu + ((u >> 16) & 1u)) >> 16);
}
__device__ __forceinline__ float bf2f(unsigned short s) {
    return __uint_as_float(((unsigned)s) << 16);
}
__device__ __forceinline__ float sigm(float x) { return 1.0f / (1.0f + __expf(-x)); }

// ---------------------------------------------------------------------------
// Prep: swizzle W_conv (fp32, [256][72][3][3]) into WMMA A-fragments (bf16).
// Fragment f = (mtile*9 + tap)*3 + kc ; within fragment: [lane 0..31][e 0..15]
// A 16x32 layout used here (contiguous K, B matches):  M = lane%16 ; K = e + 16*(lane/16)
// ---------------------------------------------------------------------------
__global__ void prep_weights(const float* __restrict__ Wc,
                             unsigned short* __restrict__ Apk) {
    int idx = blockIdx.x * 256 + threadIdx.x;          // 0 .. 221183
    if (idx >= NFRAG * 512) return;
    int e    = idx & 15;
    int lane = (idx >> 4) & 31;
    int f    = idx >> 9;
    int kc   = f % 3;
    int tap  = (f / 3) % 9;
    int mt   = f / 27;
    int M    = mt * 16 + (lane & 15);
    int K    = e + 16 * (lane >> 4);
    int kg   = kc * 32 + K;                            // padded K index 0..95
    float w = 0.0f;
    if (kg < (IN_CH + HIDDEN)) {
        int dy = tap / 3, dx = tap % 3;
        w = Wc[((M * (IN_CH + HIDDEN) + kg) * 3 + dy) * 3 + dx];
    }
    Apk[idx] = f2bf(w);
}

// zero h0 (bf16) and c (fp32)
__global__ void zero_state(unsigned short* __restrict__ h0,
                           float* __restrict__ c, int n) {
    int i = blockIdx.x * 256 + threadIdx.x;
    if (i < n) { h0[i] = 0; c[i] = 0.0f; }
}

// ---------------------------------------------------------------------------
// One ConvLSTM step. Grid = BATCH*HH workgroups (one row each), 256 threads.
// Wave (wm,wn): M in {g*64 + wm*16 .. +15} for g=0..3 ; N = wn*48 .. +47.
// ---------------------------------------------------------------------------
__global__ void __launch_bounds__(256)
conv_lstm_step(const float* __restrict__ X,            // (B,T,8,96,96)
               const unsigned short* __restrict__ Apk, // packed A fragments
               const float* __restrict__ bconv,        // (256)
               const unsigned short* __restrict__ hin, // bf16 (B,64,96,96)
               unsigned short* __restrict__ hout,      // bf16 (B,64,96,96)
               float* __restrict__ cbuf,               // fp32 (B,64,96,96)
               int t) {
    __shared__ __align__(32) unsigned short lds[3 * 98 * CPAD];  // 56,448 B

    const int blk  = blockIdx.x;
    const int b    = blk / HH;
    const int y    = blk % HH;
    const int tid  = threadIdx.x;
    const int lane = tid & 31;
    const int wave = tid >> 5;
    const int wm   = wave >> 1;     // 0..3
    const int wn   = wave & 1;      // 0..1

    // ---- stage concat(x_t, h) rows y-1..y+1, x=-1..96, 96 padded channels ----
    for (int pix = tid; pix < 3 * 98; pix += 256) {
        int row = pix / 98;
        int xx  = pix % 98 - 1;
        int gy  = y + row - 1;
        unsigned short* dst = &lds[(row * 98 + (xx + 1)) * CPAD];
        if (gy >= 0 && gy < HH && xx >= 0 && xx < WW) {
            const float* xp = X + ((((size_t)b * TSTEPS + t) * IN_CH) * HH + gy) * WW + xx;
#pragma unroll
            for (int ci = 0; ci < IN_CH; ci++) dst[ci] = f2bf(xp[(size_t)ci * NPIX]);
            const unsigned short* hp = hin + (((size_t)b * HIDDEN) * HH + gy) * WW + xx;
#pragma unroll
            for (int ci = 0; ci < HIDDEN; ci++) dst[IN_CH + ci] = hp[(size_t)ci * NPIX];
#pragma unroll
            for (int ci = IN_CH + HIDDEN; ci < CPAD; ci++) dst[ci] = 0;
        } else {
#pragma unroll
            for (int ci = 0; ci < CPAD; ci++) dst[ci] = 0;
        }
    }
    __syncthreads();

    // ---- implicit GEMM: 9 taps x 3 K-chunks, 12 WMMA tiles per wave ----
    v8f acc[4][3];
#pragma unroll
    for (int g = 0; g < 4; g++)
#pragma unroll
        for (int j = 0; j < 3; j++)
            acc[g][j] = (v8f)(0.0f);

    const int nbase = wn * 48;
    for (int tap = 0; tap < 9; tap++) {
        const int dy = tap / 3, dx = tap % 3;
        for (int kc = 0; kc < 3; kc++) {
            v16bf afrag[4];
#pragma unroll
            for (int g = 0; g < 4; g++) {
                int mt = g * 4 + wm;
                int f  = (mt * 9 + tap) * 3 + kc;
                afrag[g] = *(const v16bf*)(Apk + ((size_t)(f * 32 + lane) << 4));
            }
            v16bf bfrag[3];
#pragma unroll
            for (int j = 0; j < 3; j++) {
                int n   = nbase + j * 16 + (lane & 15);
                int idx = (dy * 98 + (n + dx)) * CPAD + kc * 32 + (lane >> 4) * 16;
                bfrag[j] = *(const v16bf*)(&lds[idx]);
            }
#pragma unroll
            for (int g = 0; g < 4; g++)
#pragma unroll
                for (int j = 0; j < 3; j++)
                    acc[g][j] = __builtin_amdgcn_wmma_f32_16x16x32_bf16(
                        false, afrag[g], false, bfrag[j],
                        (short)0, acc[g][j], false, false);
        }
    }

    // ---- wave-local LSTM pointwise epilogue ----
    // C/D tile layout: lane L, vgpr r -> M = r + 8*(L/16), N = L%16.
#pragma unroll
    for (int j = 0; j < 3; j++) {
        const int px = nbase + j * 16 + (lane & 15);
#pragma unroll
        for (int r = 0; r < 8; r++) {
            const int hc = wm * 16 + r + 8 * (lane >> 4);
            float gi = acc[0][j][r] + bconv[hc];
            float gf = acc[1][j][r] + bconv[HIDDEN + hc];
            float go = acc[2][j][r] + bconv[2 * HIDDEN + hc];
            float gg = acc[3][j][r] + bconv[3 * HIDDEN + hc];
            float i_ = sigm(gi);
            float f_ = sigm(gf);
            float o_ = sigm(go);
            float g_ = tanhf(gg);
            size_t off = (((size_t)b * HIDDEN + hc) * HH + y) * WW + px;
            float cn = f_ * cbuf[off] + i_ * g_;
            cbuf[off] = cn;
            hout[off] = f2bf(o_ * tanhf(cn));
        }
    }
}

// ---------------------------------------------------------------------------
// Decoder: out = sigmoid(1x1 conv(h) + b_dec), fp32 output (B,3,96,96)
// ---------------------------------------------------------------------------
__global__ void decoder_kernel(const unsigned short* __restrict__ h,
                               const float* __restrict__ Wd,
                               const float* __restrict__ bd,
                               float* __restrict__ out) {
    int idx = blockIdx.x * 256 + threadIdx.x;
    if (idx >= BATCH * HORIZON * NPIX) return;
    int p = idx % NPIX;
    int o = (idx / NPIX) % HORIZON;
    int b = idx / (HORIZON * NPIX);
    float s = bd[o];
    const unsigned short* hp = h + (size_t)b * HIDDEN * NPIX + p;
#pragma unroll 8
    for (int hc = 0; hc < HIDDEN; hc++)
        s += Wd[o * HIDDEN + hc] * bf2f(hp[(size_t)hc * NPIX]);
    out[idx] = sigm(s);
}

extern "C" void kernel_launch(void* const* d_in, const int* in_sizes, int n_in,
                              void* d_out, int out_size, void* d_ws, size_t ws_size,
                              hipStream_t stream) {
    (void)in_sizes; (void)n_in; (void)out_size; (void)ws_size;
    const float* X     = (const float*)d_in[0];
    const float* Wconv = (const float*)d_in[1];
    const float* bconv = (const float*)d_in[2];
    const float* Wdec  = (const float*)d_in[3];
    const float* bdec  = (const float*)d_in[4];
    float* out = (float*)d_out;

    char* ws = (char*)d_ws;
    unsigned short* Apk = (unsigned short*)(ws + A_OFF);
    unsigned short* h0  = (unsigned short*)(ws + H0_OFF);
    unsigned short* h1  = (unsigned short*)(ws + H1_OFF);
    float*          cb  = (float*)(ws + C_OFF);

    // pack weights to WMMA A-fragment layout (bf16)
    prep_weights<<<(NFRAG * 512 + 255) / 256, 256, 0, stream>>>(Wconv, Apk);
    // zero h0 and c
    zero_state<<<(HN + 255) / 256, 256, 0, stream>>>(h0, cb, HN);

    // 10 recurrent steps, ping-pong h buffers
    for (int t = 0; t < TSTEPS; t++) {
        const unsigned short* hin = (t & 1) ? h1 : h0;
        unsigned short*       hou = (t & 1) ? h0 : h1;
        conv_lstm_step<<<BATCH * HH, 256, 0, stream>>>(X, Apk, bconv, hin, hou, cb, t);
    }
    const unsigned short* hfin = (TSTEPS & 1) ? h1 : h0;  // T=10 even -> last write h0? t=9 odd writes h0
    // t=9 (odd) wrote h0, so final h is h0:
    hfin = h0;
    decoder_kernel<<<(BATCH * HORIZON * NPIX + 255) / 256, 256, 0, stream>>>(hfin, Wdec, bdec, out);
}
